// KalmanNet_26929444945955
// MI455X (gfx1250) — compile-verified, matrix operand order checked by reading, not run on hardware
//
#include <hip/hip_runtime.h>

// ---------------- problem constants ----------------
#define HDIM   1010
#define HPAD   1024
#define MDIM   10
#define TSTEPS 512
#define NSTEPS 511            // T-1 scan steps
#define SCALE_INV (1.0f/60000.0f)

// ---------------- decomposition ----------------
#define NTILES     8          // 128 batch rows / 16 (WMMA M)
#define NSPLIT     4          // workgroups cooperating per batch tile
#define WG_THREADS 256        // 8 wave32
#define NWAVES     8
#define NT_PER_WAVE 2         // 64 n-tiles / NSPLIT / NWAVES
#define KTILES     32         // 1024 / 32 (WMMA K)

// packed weight geometry: [gate(3)][ntile(64)][ktile(32)] tiles of 32 lanes x 32B
#define HALFS_PER_TILE 512
#define TILES_PER_MAT  (3*64*32)
#define HALFS_PER_MAT  ((size_t)TILES_PER_MAT * HALFS_PER_TILE)   // 3,145,728
#define BYTES_PER_MAT  (HALFS_PER_MAT * 2)                        // 6,291,456
#define GATE_HALF_STRIDE ((size_t)64*32*HALFS_PER_TILE)           // 1,048,576

typedef _Float16 v16h __attribute__((ext_vector_type(16)));
typedef float    v8f  __attribute__((ext_vector_type(8)));

__device__ __forceinline__ float sigmoidf_(float x) {
    return 1.0f / (1.0f + __expf(-x));
}

// ---------------------------------------------------------------------------
// Prologue: zero the cross-WG barrier state (ws is poisoned, not re-zeroed).
// ---------------------------------------------------------------------------
__global__ void zero_sync_kernel(unsigned* s) {
    if (threadIdx.x < 16) s[threadIdx.x] = 0u;
}

// ---------------------------------------------------------------------------
// Prologue: repack W_ih / W_hh (fp32 [3030][1010] row-major) into f16 WMMA
// B-operand tiles. One thread per packed dword (f16 pair).
// Per-lane B layout mirrors the documented 16-bit A layout (M<->N):
//   lane = n (0..15) | +16 selects K half; VGPR v holds K pair
//   k_in_tile = 2*(v&3) + 16*(v>>2) + 8*(lane>>4)
// ---------------------------------------------------------------------------
__global__ void pack_weights_kernel(const float* __restrict__ Wih,
                                    const float* __restrict__ Whh,
                                    _Float16* __restrict__ pWih,
                                    _Float16* __restrict__ pWhh) {
    const unsigned perMat = (unsigned)(HALFS_PER_MAT / 2);   // dwords per matrix
    unsigned id = blockIdx.x * blockDim.x + threadIdx.x;
    if (id >= 2u * perMat) return;
    const unsigned mat = id / perMat;
    const unsigned r   = id % perMat;
    const unsigned v    = r & 7u;
    const unsigned lane = (r >> 3) & 31u;
    const unsigned kt   = (r >> 8) & 31u;
    const unsigned nt   = (r >> 13) & 63u;
    const unsigned g    = r >> 19;                           // 0..2 (gate)
    const unsigned n    = nt * 16u + (lane & 15u);
    const unsigned k    = kt * 32u + 2u*(v & 3u) + 16u*(v >> 2) + 8u*(lane >> 4);
    const float* W = mat ? Whh : Wih;
    float a = 0.0f, b = 0.0f;
    if (n < HDIM) {
        const size_t row = (size_t)(g * HDIM + n) * HDIM;
        if (k < HDIM)      a = W[row + k];
        if (k + 1u < HDIM) b = W[row + k + 1u];
    }
    _Float16* dst = (mat ? pWhh : pWih) + (size_t)r * 2u;
    dst[0] = (_Float16)a;
    dst[1] = (_Float16)b;
}

// ---------------------------------------------------------------------------
// Persistent recurrent kernel: 32 WGs = 8 batch tiles x 4 HDIM splits.
// ---------------------------------------------------------------------------
__launch_bounds__(WG_THREADS)
__global__ void kalman_persistent_kernel(
    const float* __restrict__ vs,      // [128][512]
    const float* __restrict__ iapp,    // [128]
    const float* __restrict__ state0,  // [128][10]
    const float* __restrict__ Win,     // [1010][11]
    const float* __restrict__ bin,     // [1010]
    const float* __restrict__ bih,     // [3030]
    const float* __restrict__ bhh,     // [3030]
    const float* __restrict__ Wout,    // [10][1010]
    const float* __restrict__ bout,    // [10]
    const _Float16* __restrict__ pWih, // packed
    const _Float16* __restrict__ pWhh, // packed
    float* __restrict__ hbuf,          // [2][NTILES][16][HPAD] f32
    unsigned* __restrict__ syncArea,   // cnt[8] | phase[8]
    float* __restrict__ out)           // [128][511][10]
{
    __shared__ float sX[16 * HPAD];       // x (A operand), f32
    __shared__ float sH[16 * HPAD];       // h (A operand + GRU old-h), f32
    __shared__ float sRed[16 * 16 * MDIM];
    __shared__ float sSt[16 * MDIM], sPred[16 * MDIM], sF4[16 * MDIM];
    __shared__ float sFeat[16 * 12];
    __shared__ float sF2[16];

    const int tid   = threadIdx.x;
    const int lane  = tid & 31;
    const int wave  = tid >> 5;
    const int tile  = blockIdx.x / NSPLIT;
    const int split = blockIdx.x % NSPLIT;
    const int nt0   = split * 16 + wave * NT_PER_WAVE;
    const int mrow  = lane & 15;
    const int half  = lane >> 4;

    unsigned* cnt   = syncArea;
    unsigned* phase = syncArea + 8;

    // ---- init: h = 0, state = state0, f4 = 0 ----
    for (int i = tid; i < 16 * HPAD; i += WG_THREADS) sH[i] = 0.0f;
    for (int i = tid; i < 16 * MDIM; i += WG_THREADS) {
        const int m = i / MDIM, j = i % MDIM;
        sSt[i] = state0[(tile * 16 + m) * MDIM + j];
        sF4[i] = 0.0f;
    }
    __syncthreads();

    for (int step = 0; step < NSTEPS; ++step) {
        // ===== phase A (redundant per split): Euler predict, f2, feats =====
        if (tid < 16) {
            const int m = tid;
            const int b = tile * 16 + m;
            const float v   = sSt[m*MDIM+0], n   = sSt[m*MDIM+1];
            const float gl  = sSt[m*MDIM+2], gk  = sSt[m*MDIM+3];
            const float gca = sSt[m*MDIM+4], phi = sSt[m*MDIM+5];
            const float v1  = sSt[m*MDIM+6], v2  = sSt[m*MDIM+7];
            const float v3  = sSt[m*MDIM+8], v4  = sSt[m*MDIM+9];
            const float il   = gl * (v + 60.0f);
            const float ik   = gk * n * (v + 84.0f);
            const float minf = 0.5f * (1.0f + tanhf((v - v1) / v2));
            const float ica  = gca * minf * (v - 120.0f);
            const float fv   = (iapp[b] - il - ik - ica) * 0.05f;   // /C, C=20
            const float ninf = 0.5f * (1.0f + tanhf((v - v3) / v4));
            const float tau  = 1.0f / coshf((v - v3) / (2.0f * v4));
            const float fn   = phi * (ninf - n) / tau;
            const float vn   = v + 0.05f * fv;
            const float nn   = fminf(fmaxf(n + 0.05f * fn, 0.0f), 1.0f);
            sPred[m*MDIM+0] = vn;
            sPred[m*MDIM+1] = nn;
            #pragma unroll
            for (int j = 2; j < MDIM; ++j) sPred[m*MDIM+j] = sSt[m*MDIM+j];
            const float f2 = vs[(size_t)b * TSTEPS + step + 1] - vn;
            sF2[m] = f2;
            #pragma unroll
            for (int j = 0; j < MDIM; ++j) sFeat[m*12+j] = sF4[m*MDIM+j] * SCALE_INV;
            sFeat[m*12+10] = f2 * SCALE_INV;
        }
        __syncthreads();

        // x = feats @ Win^T + bin  (tiny: K=11), padded to HPAD with zeros
        for (int c = tid; c < HPAD; c += WG_THREADS) {
            if (c < HDIM) {
                float w[11];
                #pragma unroll
                for (int i = 0; i < 11; ++i) w[i] = Win[c * 11 + i];
                const float bb = bin[c];
                for (int m = 0; m < 16; ++m) {
                    float acc = bb;
                    #pragma unroll
                    for (int i = 0; i < 11; ++i) acc += sFeat[m*12+i] * w[i];
                    sX[m * HPAD + c] = acc;
                }
            } else {
                for (int m = 0; m < 16; ++m) sX[m * HPAD + c] = 0.0f;
            }
        }
        __syncthreads();

        // ===== phase B: fused gi/gh GEMM via WMMA f32<-f16, K=1024 =====
        const v8f vzero = {0.f,0.f,0.f,0.f,0.f,0.f,0.f,0.f};
        v8f accR[NT_PER_WAVE], accZ[NT_PER_WAVE], accIN[NT_PER_WAVE], accHN[NT_PER_WAVE];
        #pragma unroll
        for (int t2 = 0; t2 < NT_PER_WAVE; ++t2) {
            accR[t2] = vzero; accZ[t2] = vzero; accIN[t2] = vzero; accHN[t2] = vzero;
        }

        for (int kt = 0; kt < KTILES; ++kt) {
            // gather A operands (x and h) from LDS f32 -> f16 per WMMA A layout
            v16h ax, ah;
            #pragma unroll
            for (int v = 0; v < 8; ++v) {
                const int k = kt * 32 + 2*(v & 3) + 16*(v >> 2) + 8*half;
                const float2 fx = *(const float2*)&sX[mrow * HPAD + k];
                const float2 fh = *(const float2*)&sH[mrow * HPAD + k];
                ax[2*v]   = (_Float16)fx.x;  ax[2*v+1] = (_Float16)fx.y;
                ah[2*v]   = (_Float16)fh.x;  ah[2*v+1] = (_Float16)fh.y;
            }
            #pragma unroll
            for (int t2 = 0; t2 < NT_PER_WAVE; ++t2) {
                const int nt = nt0 + t2;
                const size_t base = ((size_t)nt * 32 + kt) * HALFS_PER_TILE + (size_t)lane * 16;
                const v16h bRi = *(const v16h*)(pWih + base);
                const v16h bRh = *(const v16h*)(pWhh + base);
                const v16h bZi = *(const v16h*)(pWih + base + GATE_HALF_STRIDE);
                const v16h bZh = *(const v16h*)(pWhh + base + GATE_HALF_STRIDE);
                const v16h bNi = *(const v16h*)(pWih + base + 2*GATE_HALF_STRIDE);
                const v16h bNh = *(const v16h*)(pWhh + base + 2*GATE_HALF_STRIDE);
                accR[t2]  = __builtin_amdgcn_wmma_f32_16x16x32_f16(false, ax, false, bRi, (short)0, accR[t2],  false, false);
                accR[t2]  = __builtin_amdgcn_wmma_f32_16x16x32_f16(false, ah, false, bRh, (short)0, accR[t2],  false, false);
                accZ[t2]  = __builtin_amdgcn_wmma_f32_16x16x32_f16(false, ax, false, bZi, (short)0, accZ[t2],  false, false);
                accZ[t2]  = __builtin_amdgcn_wmma_f32_16x16x32_f16(false, ah, false, bZh, (short)0, accZ[t2],  false, false);
                accIN[t2] = __builtin_amdgcn_wmma_f32_16x16x32_f16(false, ax, false, bNi, (short)0, accIN[t2], false, false);
                accHN[t2] = __builtin_amdgcn_wmma_f32_16x16x32_f16(false, ah, false, bNh, (short)0, accHN[t2], false, false);
            }
            if (kt + 1 < KTILES) {  // pull next kt's weight tiles toward L0/L2
                const size_t nb = ((size_t)nt0 * 32 + kt + 1) * HALFS_PER_TILE + (size_t)lane * 16;
                __builtin_prefetch(pWih + nb, 0, 1);
                __builtin_prefetch(pWhh + nb, 0, 1);
            }
        }

        // ===== GRU elementwise in registers, publish h_new slice to global =====
        const int parity = step & 1;
        float* hb = hbuf + ((size_t)parity * NTILES + tile) * 16 * HPAD;
        #pragma unroll
        for (int t2 = 0; t2 < NT_PER_WAVE; ++t2) {
            const int n = (nt0 + t2) * 16 + mrow;
            const bool valid = (n < HDIM);
            const float bR = valid ? (bih[n] + bhh[n]) : 0.0f;
            const float bZ = valid ? (bih[HDIM + n] + bhh[HDIM + n]) : 0.0f;
            const float bI = valid ? bih[2 * HDIM + n] : 0.0f;
            const float bH = valid ? bhh[2 * HDIM + n] : 0.0f;
            #pragma unroll
            for (int v = 0; v < 8; ++v) {
                const int m = v + 8 * half;                 // C/D layout: M = v (+8 hi half)
                const float r  = sigmoidf_(accR[t2][v] + bR);
                const float z  = sigmoidf_(accZ[t2][v] + bZ);
                const float ng = tanhf(accIN[t2][v] + bI + r * (accHN[t2][v] + bH));
                const float ho = sH[m * HPAD + n];
                hb[m * HPAD + n] = (1.0f - z) * ng + z * ho;
            }
        }

        // ===== 4-way split barrier per tile (monotonic phase) =====
        __threadfence();
        __syncthreads();
        if (tid == 0) {
            const unsigned old = __hip_atomic_fetch_add(&cnt[tile], 1u,
                                   __ATOMIC_ACQ_REL, __HIP_MEMORY_SCOPE_AGENT);
            if (old == (unsigned)((step + 1) * NSPLIT - 1)) {
                __hip_atomic_store(&phase[tile], (unsigned)(step + 1),
                                   __ATOMIC_RELEASE, __HIP_MEMORY_SCOPE_AGENT);
            } else {
                while (__hip_atomic_load(&phase[tile], __ATOMIC_ACQUIRE,
                                         __HIP_MEMORY_SCOPE_AGENT) < (unsigned)(step + 1)) {
                    __builtin_amdgcn_s_sleep(2);
                }
            }
        }
        __syncthreads();
        __threadfence();

        // pull the fully-assembled h_{t} into LDS (A operand for next step + KG)
        for (int i = tid; i < 16 * HPAD / 4; i += WG_THREADS)
            ((float4*)sH)[i] = ((const float4*)hb)[i];
        __syncthreads();

        // ===== KG = h_new @ Wout^T + bout (redundant), state update =====
        {
            const int m = tid >> 4, s = tid & 15;
            float a[MDIM];
            #pragma unroll
            for (int j = 0; j < MDIM; ++j) a[j] = 0.0f;
            for (int k = s; k < HDIM; k += 16) {
                const float hv = sH[m * HPAD + k];
                #pragma unroll
                for (int j = 0; j < MDIM; ++j) a[j] += hv * Wout[j * HDIM + k];
            }
            #pragma unroll
            for (int j = 0; j < MDIM; ++j) sRed[(m * 16 + s) * MDIM + j] = a[j];
        }
        __syncthreads();
        if (tid < 16 * MDIM) {
            const int m = tid / MDIM, j = tid % MDIM;
            float kg = bout[j];
            #pragma unroll
            for (int s = 0; s < 16; ++s) kg += sRed[(m * 16 + s) * MDIM + j];
            float stv = sPred[m * MDIM + j] + kg * sF2[m];
            if (j == 1) stv = fminf(fmaxf(stv, 0.0f), 1.0f);
            sSt[m * MDIM + j] = stv;
            sF4[m * MDIM + j] = stv - sPred[m * MDIM + j];
            if (split == 0)
                out[((size_t)(tile * 16 + m) * NSTEPS + step) * MDIM + j] = stv;
        }
        __syncthreads();
    }
}

// ---------------------------------------------------------------------------
extern "C" void kernel_launch(void* const* d_in, const int* in_sizes, int n_in,
                              void* d_out, int out_size, void* d_ws, size_t ws_size,
                              hipStream_t stream) {
    const float* vs    = (const float*)d_in[0];
    const float* iapp  = (const float*)d_in[1];
    const float* st0   = (const float*)d_in[2];
    const float* Win   = (const float*)d_in[3];
    const float* bin   = (const float*)d_in[4];
    const float* Wih   = (const float*)d_in[5];
    const float* bih   = (const float*)d_in[6];
    const float* Whh   = (const float*)d_in[7];
    const float* bhh   = (const float*)d_in[8];
    const float* Wout  = (const float*)d_in[9];
    const float* bout  = (const float*)d_in[10];

    char* ws = (char*)d_ws;
    _Float16* pWih = (_Float16*)(ws);
    _Float16* pWhh = (_Float16*)(ws + BYTES_PER_MAT);
    float*    hbuf = (float*)(ws + 2 * BYTES_PER_MAT);
    const size_t hbufBytes = (size_t)2 * NTILES * 16 * HPAD * sizeof(float);
    unsigned* syncArea = (unsigned*)(ws + 2 * BYTES_PER_MAT + hbufBytes);

    zero_sync_kernel<<<1, 32, 0, stream>>>(syncArea);

    const unsigned packThreads = 2u * (unsigned)(HALFS_PER_MAT / 2);
    pack_weights_kernel<<<(packThreads + 255) / 256, 256, 0, stream>>>(
        Wih, Whh, pWih, pWhh);

    kalman_persistent_kernel<<<NTILES * NSPLIT, WG_THREADS, 0, stream>>>(
        vs, iapp, st0, Win, bin, bih, bhh, Wout, bout,
        pWih, pWhh, hbuf, syncArea, (float*)d_out);
}
